// EnhancedGCN_1906965479692
// MI455X (gfx1250) — matrix operand order repaired and checked by reading, not compiled, not run on hardware
//
#include <hip/hip_runtime.h>
#include <hip/hip_fp16.h>

// ---------------- model constants (from reference) ----------------
#define F_IN   128
#define HID    16
#define HEADS  12
#define OUTC   32

typedef __attribute__((ext_vector_type(16))) _Float16 v16h;
typedef __attribute__((ext_vector_type(8)))  float    v8f;

// ============================================================
// small utility kernels
// ============================================================
__global__ void fill_f32(float* p, float v, int n) {
    int i = blockIdx.x * blockDim.x + threadIdx.x;
    if (i < n) p[i] = v;
}

// float atomic-max via int-max / uint-min bit trick (correct for all signs)
__device__ __forceinline__ void atomicMaxFloat(float* addr, float val) {
    if (val >= 0.0f) atomicMax((int*)addr, __float_as_int(val));
    else             atomicMin((unsigned int*)addr, __float_as_uint(val));
}

// ============================================================
// Operand packing into CDNA5 WMMA fragment layouts (f32 -> f16,
// zero-padded K). One thread per packed f16 element.
//
// A fragment layout (16-bit A 16x32, ISA 7.12.2):
//   lane<16 : row=lane,    K = e (e<8) / e+8 (e>=8)          -> 0..7,16..23
//   lane>=16: row=lane-16, K = e+8 (e<8) / e+16 (e>=8)       -> 8..15,24..31
// Packed index: ((rt*ktiles + kt)*32 + lane)*16 + e
// ============================================================
__global__ void packA_k(const float* __restrict__ A, _Float16* __restrict__ Ap,
                        int M, int K, int ktiles, int total) {
    int i = blockIdx.x * blockDim.x + threadIdx.x;
    if (i >= total) return;
    int e    = i & 15;
    int lane = (i >> 4) & 31;
    int kt   = (i >> 9) % ktiles;
    int rt   = (i >> 9) / ktiles;
    int hi   = lane >> 4;
    int row  = rt * 16 + (lane & 15);
    int k    = kt * 32 + ((e < 8) ? (e + hi * 8) : (e + 8 + hi * 8));
    Ap[i] = (row < M && k < K) ? (_Float16)A[(size_t)row * K + k] : (_Float16)0.0f;
}

// B fragment layout (16-bit B 32x16): bv[e] = B[k0 + (e<8 ? e : e+8) + 8*(lane>=16)][col],
// col = lane&15.  Packed index: ((kt*ctiles + ct)*32 + lane)*16 + e
__global__ void packB_k(const float* __restrict__ B, _Float16* __restrict__ Bp,
                        int K, int N, int ctiles, int total) {
    int i = blockIdx.x * blockDim.x + threadIdx.x;
    if (i >= total) return;
    int e    = i & 15;
    int lane = (i >> 4) & 31;
    int ct   = (i >> 9) % ctiles;
    int kt   = (i >> 9) / ctiles;
    int kr   = kt * 32 + ((e < 8) ? e : (e + 8)) + ((lane >> 4) ? 8 : 0);
    int col  = ct * 16 + (lane & 15);
    Bp[i] = (kr < K && col < N) ? (_Float16)B[(size_t)kr * N + col] : (_Float16)0.0f;
}

// ============================================================
// WMMA GEMM on pre-packed fragments. One wave32 per 16x16 output
// tile. Inner loop: two 32B vector loads + one v_wmma, no branches.
// ============================================================
__global__ __launch_bounds__(32)
void wmma_gemm_k(const _Float16* __restrict__ Ap, const _Float16* __restrict__ Bp,
                 float* __restrict__ C, int M, int N, int ktiles) {
    const int t  = threadIdx.x;
    const int lr = t & 15;
    const int hi = t >> 4;
    const int rt = blockIdx.x;
    const int ct = blockIdx.y;
    const int ctiles = gridDim.y;

    v8f acc = {};
    const _Float16* ap = Ap + ((size_t)rt * ktiles * 32 + t) * 16;
    const _Float16* bp = Bp + ((size_t)ct * 32 + t) * 16;
    const size_t bstride = (size_t)ctiles * 32 * 16;

    for (int kt = 0; kt < ktiles; ++kt) {
        v16h av = *(const v16h*)(ap + (size_t)kt * 512);
        v16h bv = *(const v16h*)(bp + (size_t)kt * bstride);
        acc = __builtin_amdgcn_wmma_f32_16x16x32_f16(
                  false, av, false, bv, (short)0, acc, false, false);
    }

    // C/D layout: VGPR r -> row rt*16 + r + 8*hi, col = ct*16 + (lane&15)
    const int col = ct * 16 + lr;
    const int rb  = rt * 16 + hi * 8;
    if (rb + 8 <= M) {
        #pragma unroll
        for (int r = 0; r < 8; ++r) C[(size_t)(rb + r) * N + col] = acc[r];
    } else {
        #pragma unroll
        for (int r = 0; r < 8; ++r)
            if (rb + r < M) C[(size_t)(rb + r) * N + col] = acc[r];
    }
}

// ============================================================
// GCN pieces
// ============================================================
__global__ void degree_k(const int* __restrict__ ei, float* deg, int E, int N) {
    int e = blockIdx.x * blockDim.x + threadIdx.x;
    int tot = E + N;
    if (e >= tot) return;
    int d = (e < E) ? ei[E + e] : (e - E);       // dst row of edge_index, or self-loop
    unsafeAtomicAdd(&deg[d], 1.0f);
}

__global__ void dinv_k(float* deg, int N) {
    int i = blockIdx.x * blockDim.x + threadIdx.x;
    if (i >= N) return;
    float d = deg[i];
    deg[i] = (d > 0.0f) ? rsqrtf(d) : 0.0f;
}

// out[dst] += h[src] * dinv[src]*dinv[dst], one thread per edge
__global__ void gcn_scatter_k(const float* __restrict__ h, const int* __restrict__ ei,
                              const float* __restrict__ dinv, float* __restrict__ out,
                              int E, int N, int C) {
    int e = blockIdx.x * blockDim.x + threadIdx.x;
    int tot = E + N;
    if (e >= tot) return;
    int s, d;
    if (e < E) { s = ei[e]; d = ei[E + e]; } else { s = d = e - E; }
    float w = dinv[s] * dinv[d];
    const float* hs = h   + (size_t)s * C;
    float*       od = out + (size_t)d * C;
    for (int f = 0; f < C; ++f) unsafeAtomicAdd(&od[f], hs[f] * w);
}

// in-place: x = relu(x + bias[c])
__global__ void gcn_finalize_relu_k(float* x, const float* __restrict__ bias, int N, int C) {
    int i = blockIdx.x * blockDim.x + threadIdx.x;
    if (i >= N * C) return;
    float v = x[i] + bias[i % C];
    x[i] = v > 0.0f ? v : 0.0f;
}

// out = relu(agg + bias[c]) + res   (residual GCN-3)
__global__ void gcn_finalize_relu_res_k(const float* __restrict__ agg, const float* __restrict__ bias,
                                        const float* __restrict__ res, float* __restrict__ out,
                                        int N, int C) {
    int i = blockIdx.x * blockDim.x + threadIdx.x;
    if (i >= N * C) return;
    float v = agg[i] + bias[i % C];
    v = v > 0.0f ? v : 0.0f;
    out[i] = v + res[i];
}

// ============================================================
// GAT pieces
// ============================================================
__global__ void gat_scores_k(const float* __restrict__ hg, const float* __restrict__ att_s,
                             const float* __restrict__ att_d, float* __restrict__ a_s,
                             float* __restrict__ a_d, int N) {
    int i = blockIdx.x * blockDim.x + threadIdx.x;
    if (i >= N * HEADS) return;
    int n = i / HEADS, h = i % HEADS;
    const float* hv = hg + (size_t)n * (HEADS * HID) + h * HID;
    float ss = 0.0f, sd = 0.0f;
    #pragma unroll
    for (int c = 0; c < HID; ++c) {
        ss += hv[c] * att_s[h * HID + c];
        sd += hv[c] * att_d[h * HID + c];
    }
    a_s[i] = ss;
    a_d[i] = sd;
}

__device__ __forceinline__ void gat_edge(const int* ei, int E, int N, int idx,
                                         int& s, int& d, int& h) {
    int e = idx / HEADS;
    h = idx % HEADS;
    if (e < E) { s = ei[e]; d = ei[E + e]; } else { s = d = e - E; }
}

__global__ void gat_max_k(const int* __restrict__ ei, const float* __restrict__ a_s,
                          const float* __restrict__ a_d, float* __restrict__ mmax,
                          int E, int N) {
    int idx = blockIdx.x * blockDim.x + threadIdx.x;
    if (idx >= (E + N) * HEADS) return;
    int s, d, h; gat_edge(ei, E, N, idx, s, d, h);
    float e = a_s[s * HEADS + h] + a_d[d * HEADS + h];
    e = e > 0.0f ? e : 0.2f * e;                       // leaky_relu(0.2)
    atomicMaxFloat(&mmax[d * HEADS + h], e);
}

__global__ void gat_denom_k(const int* __restrict__ ei, const float* __restrict__ a_s,
                            const float* __restrict__ a_d, const float* __restrict__ mmax,
                            float* __restrict__ denom, int E, int N) {
    int idx = blockIdx.x * blockDim.x + threadIdx.x;
    if (idx >= (E + N) * HEADS) return;
    int s, d, h; gat_edge(ei, E, N, idx, s, d, h);
    float e = a_s[s * HEADS + h] + a_d[d * HEADS + h];
    e = e > 0.0f ? e : 0.2f * e;
    unsafeAtomicAdd(&denom[d * HEADS + h], __expf(e - mmax[d * HEADS + h]));
}

__global__ void gat_scatter_k(const int* __restrict__ ei, const float* __restrict__ a_s,
                              const float* __restrict__ a_d, const float* __restrict__ mmax,
                              const float* __restrict__ denom, const float* __restrict__ hg,
                              float* __restrict__ out, int E, int N) {
    int idx = blockIdx.x * blockDim.x + threadIdx.x;
    if (idx >= (E + N) * HEADS) return;
    int s, d, h; gat_edge(ei, E, N, idx, s, d, h);
    float e = a_s[s * HEADS + h] + a_d[d * HEADS + h];
    e = e > 0.0f ? e : 0.2f * e;
    float den   = denom[d * HEADS + h];
    float alpha = __expf(e - mmax[d * HEADS + h]) / (den > 0.0f ? den : 1.0f);
    const float* hv = hg  + (size_t)s * (HEADS * HID) + h * HID;
    float*       ov = out + (size_t)d * (HEADS * HID) + h * HID;
    #pragma unroll
    for (int c = 0; c < HID; ++c) unsafeAtomicAdd(&ov[c], alpha * hv[c]);
}

// in-place: x = elu(x + bg[c])
__global__ void gat_finalize_elu_k(float* x, const float* __restrict__ bg, int N) {
    int i = blockIdx.x * blockDim.x + threadIdx.x;
    if (i >= N * HEADS * HID) return;
    float v = x[i] + bg[i % (HEADS * HID)];
    x[i] = v > 0.0f ? v : (__expf(v) - 1.0f);
}

// ============================================================
// readout
// ============================================================
__global__ void mean_k(const float* __restrict__ x4, float* __restrict__ mean, int N) {
    int i = blockIdx.x * blockDim.x + threadIdx.x;
    if (i >= N * HID) return;
    unsafeAtomicAdd(&mean[i % HID], x4[i] * (1.0f / (float)N));
}

__global__ __launch_bounds__(32)
void final_linear_k(const float* __restrict__ mean, const float* __restrict__ Wlin,
                    const float* __restrict__ blin, float* __restrict__ out) {
    int o = threadIdx.x;
    if (o >= OUTC) return;
    float acc = blin[o];
    #pragma unroll
    for (int k = 0; k < HID; ++k) acc += mean[k] * Wlin[k * OUTC + o];
    out[o] = acc;
}

// ============================================================
// host launcher
// ============================================================
extern "C" void kernel_launch(void* const* d_in, const int* in_sizes, int n_in,
                              void* d_out, int out_size, void* d_ws, size_t ws_size,
                              hipStream_t stream) {
    const float* x        = (const float*)d_in[0];
    const int*   ei       = (const int*)  d_in[1];
    const float* W1       = (const float*)d_in[2];
    const float* b1       = (const float*)d_in[3];
    const float* Wg       = (const float*)d_in[4];
    const float* att_src  = (const float*)d_in[5];
    const float* att_dst  = (const float*)d_in[6];
    const float* bg       = (const float*)d_in[7];
    const float* W2       = (const float*)d_in[8];
    const float* b2       = (const float*)d_in[9];
    const float* W3       = (const float*)d_in[10];
    const float* b3       = (const float*)d_in[11];
    const float* Wlin     = (const float*)d_in[12];
    const float* blin     = (const float*)d_in[13];
    float*       out      = (float*)d_out;

    const int N = in_sizes[0] / F_IN;   // 25000
    const int E = in_sizes[1] / 2;      // 400000
    const int Etot = E + N;
    const int GC = HEADS * HID;         // 192

    const int rtiles = (N + 15) / 16;   // 1563
    const int kt1 = F_IN / 32;          // 4   (K=128)
    const int ktg = 1;                  // K=16 -> padded to 32
    const int kt2 = GC / 32;            // 6   (K=192)
    const int kt3 = 1;                  // K=16 -> padded to 32

    // ---- carve workspace ----
    char* ws = (char*)d_ws;
    size_t off = 0;
    auto carve = [&](size_t bytes) -> void* {
        void* p = ws + off;
        off = (off + bytes + 255) & ~(size_t)255;
        return p;
    };
    float*    dinv  = (float*)carve((size_t)N * sizeof(float));
    // packed A fragments
    _Float16* xp    = (_Float16*)carve((size_t)rtiles * kt1 * 512 * sizeof(_Float16));
    _Float16* x1p   = (_Float16*)carve((size_t)rtiles * ktg * 512 * sizeof(_Float16));
    _Float16* x2p   = (_Float16*)carve((size_t)rtiles * kt2 * 512 * sizeof(_Float16));
    _Float16* x3p   = (_Float16*)carve((size_t)rtiles * kt3 * 512 * sizeof(_Float16));
    // packed B fragments
    _Float16* W1p   = (_Float16*)carve((size_t)kt1 * 1     * 512 * sizeof(_Float16));
    _Float16* Wgp   = (_Float16*)carve((size_t)ktg * HEADS * 512 * sizeof(_Float16));
    _Float16* W2p   = (_Float16*)carve((size_t)kt2 * 1     * 512 * sizeof(_Float16));
    _Float16* W3p   = (_Float16*)carve((size_t)kt3 * 1     * 512 * sizeof(_Float16));
    // activations
    float*    h1    = (float*)carve((size_t)N * HID * sizeof(float));
    float*    x1    = (float*)carve((size_t)N * HID * sizeof(float));
    float*    hg    = (float*)carve((size_t)N * GC  * sizeof(float));
    float*    a_s   = (float*)carve((size_t)N * HEADS * sizeof(float));
    float*    a_d   = (float*)carve((size_t)N * HEADS * sizeof(float));
    float*    mmax  = (float*)carve((size_t)N * HEADS * sizeof(float));
    float*    denom = (float*)carve((size_t)N * HEADS * sizeof(float));
    float*    x2    = (float*)carve((size_t)N * GC  * sizeof(float));  // gat agg -> elu in place
    float*    h2    = (float*)carve((size_t)N * HID * sizeof(float));
    float*    x3    = (float*)carve((size_t)N * HID * sizeof(float));  // agg2 -> relu in place
    float*    h3    = (float*)carve((size_t)N * HID * sizeof(float));
    float*    agg3  = (float*)carve((size_t)N * HID * sizeof(float));
    float*    x4    = (float*)carve((size_t)N * HID * sizeof(float));
    float*    meanb = (float*)carve(HID * sizeof(float));
    (void)ws_size; (void)n_in; (void)out_size;

    const int TB = 256;
    auto blks = [&](int n) { return (n + TB - 1) / TB; };

    // ---- degrees / symmetric norm ----
    hipLaunchKernelGGL(fill_f32, dim3(blks(N)), dim3(TB), 0, stream, dinv, 0.0f, N);
    hipLaunchKernelGGL(degree_k, dim3(blks(Etot)), dim3(TB), 0, stream, ei, dinv, E, N);
    hipLaunchKernelGGL(dinv_k,   dim3(blks(N)), dim3(TB), 0, stream, dinv, N);

    // ---- pack weights into B fragments (tiny, reused by every tile) ----
    int tW1 = kt1 * 1 * 512, tWg = ktg * HEADS * 512, tW2 = kt2 * 1 * 512, tW3 = kt3 * 1 * 512;
    hipLaunchKernelGGL(packB_k, dim3(blks(tW1)), dim3(TB), 0, stream, W1, W1p, F_IN, HID, 1, tW1);
    hipLaunchKernelGGL(packB_k, dim3(blks(tWg)), dim3(TB), 0, stream, Wg, Wgp, HID, GC, HEADS, tWg);
    hipLaunchKernelGGL(packB_k, dim3(blks(tW2)), dim3(TB), 0, stream, W2, W2p, GC, HID, 1, tW2);
    hipLaunchKernelGGL(packB_k, dim3(blks(tW3)), dim3(TB), 0, stream, W3, W3p, HID, HID, 1, tW3);

    // ---- GCN-1: h1 = x @ W1 (WMMA, K=128) ; aggregate ; relu(+b1) ----
    int tA1 = rtiles * kt1 * 512;
    hipLaunchKernelGGL(packA_k, dim3(blks(tA1)), dim3(TB), 0, stream, x, xp, N, F_IN, kt1, tA1);
    hipLaunchKernelGGL(wmma_gemm_k, dim3(rtiles, 1), dim3(32), 0, stream, xp, W1p, h1, N, HID, kt1);
    hipLaunchKernelGGL(fill_f32, dim3(blks(N*HID)), dim3(TB), 0, stream, x1, 0.0f, N*HID);
    hipLaunchKernelGGL(gcn_scatter_k, dim3(blks(Etot)), dim3(TB), 0, stream, h1, ei, dinv, x1, E, N, HID);
    hipLaunchKernelGGL(gcn_finalize_relu_k, dim3(blks(N*HID)), dim3(TB), 0, stream, x1, b1, N, HID);

    // ---- GAT: hg = x1 @ Wg (WMMA, K=16 padded, 12 col-tiles) ----
    int tAg = rtiles * ktg * 512;
    hipLaunchKernelGGL(packA_k, dim3(blks(tAg)), dim3(TB), 0, stream, x1, x1p, N, HID, ktg, tAg);
    hipLaunchKernelGGL(wmma_gemm_k, dim3(rtiles, HEADS), dim3(32), 0, stream, x1p, Wgp, hg, N, GC, ktg);
    hipLaunchKernelGGL(gat_scores_k, dim3(blks(N*HEADS)), dim3(TB), 0, stream, hg, att_src, att_dst, a_s, a_d, N);
    hipLaunchKernelGGL(fill_f32, dim3(blks(N*HEADS)), dim3(TB), 0, stream, mmax, -3.0e38f, N*HEADS);
    hipLaunchKernelGGL(gat_max_k, dim3(blks(Etot*HEADS)), dim3(TB), 0, stream, ei, a_s, a_d, mmax, E, N);
    hipLaunchKernelGGL(fill_f32, dim3(blks(N*HEADS)), dim3(TB), 0, stream, denom, 0.0f, N*HEADS);
    hipLaunchKernelGGL(gat_denom_k, dim3(blks(Etot*HEADS)), dim3(TB), 0, stream, ei, a_s, a_d, mmax, denom, E, N);
    hipLaunchKernelGGL(fill_f32, dim3(blks(N*GC)), dim3(TB), 0, stream, x2, 0.0f, N*GC);
    hipLaunchKernelGGL(gat_scatter_k, dim3(blks(Etot*HEADS)), dim3(TB), 0, stream,
                       ei, a_s, a_d, mmax, denom, hg, x2, E, N);
    hipLaunchKernelGGL(gat_finalize_elu_k, dim3(blks(N*GC)), dim3(TB), 0, stream, x2, bg, N);

    // ---- GCN-2: h2 = x2 @ W2 (WMMA, K=192) ; aggregate ; relu(+b2) -> x3 ----
    int tA2 = rtiles * kt2 * 512;
    hipLaunchKernelGGL(packA_k, dim3(blks(tA2)), dim3(TB), 0, stream, x2, x2p, N, GC, kt2, tA2);
    hipLaunchKernelGGL(wmma_gemm_k, dim3(rtiles, 1), dim3(32), 0, stream, x2p, W2p, h2, N, HID, kt2);
    hipLaunchKernelGGL(fill_f32, dim3(blks(N*HID)), dim3(TB), 0, stream, x3, 0.0f, N*HID);
    hipLaunchKernelGGL(gcn_scatter_k, dim3(blks(Etot)), dim3(TB), 0, stream, h2, ei, dinv, x3, E, N, HID);
    hipLaunchKernelGGL(gcn_finalize_relu_k, dim3(blks(N*HID)), dim3(TB), 0, stream, x3, b2, N, HID);

    // ---- GCN-3: h3 = x3 @ W3 (WMMA, K=16 padded) ; aggregate ; relu(+b3) + x3 -> x4 ----
    int tA3 = rtiles * kt3 * 512;
    hipLaunchKernelGGL(packA_k, dim3(blks(tA3)), dim3(TB), 0, stream, x3, x3p, N, HID, kt3, tA3);
    hipLaunchKernelGGL(wmma_gemm_k, dim3(rtiles, 1), dim3(32), 0, stream, x3p, W3p, h3, N, HID, kt3);
    hipLaunchKernelGGL(fill_f32, dim3(blks(N*HID)), dim3(TB), 0, stream, agg3, 0.0f, N*HID);
    hipLaunchKernelGGL(gcn_scatter_k, dim3(blks(Etot)), dim3(TB), 0, stream, h3, ei, dinv, agg3, E, N, HID);
    hipLaunchKernelGGL(gcn_finalize_relu_res_k, dim3(blks(N*HID)), dim3(TB), 0, stream, agg3, b3, x3, x4, N, HID);

    // ---- mean over nodes + final linear ----
    hipLaunchKernelGGL(fill_f32, dim3(1), dim3(HID), 0, stream, meanb, 0.0f, HID);
    hipLaunchKernelGGL(mean_k, dim3(blks(N*HID)), dim3(TB), 0, stream, x4, meanb, N);
    hipLaunchKernelGGL(final_linear_k, dim3(1), dim3(32), 0, stream, meanb, Wlin, blin, out);
}
